// Cell_B_10746008174947
// MI455X (gfx1250) — matrix operand order repaired and checked by reading, not compiled, not run on hardware
//
#include <hip/hip_runtime.h>
#include <stdint.h>

// ---- problem constants ----
#define L_    2048
#define KCH   64      // input channels
#define OC1   512     // CB*AB
#define OC2   256     // CSB*ASB
#define IC2   512
#define LT    64      // L columns per workgroup tile
#define EXT   80      // extended tile (halo 8 each side)
#define XLD   72      // LDS row stride (ushorts): 144B rows -> 16B aligned, bank-spread
#define VLD   264     // vtile row stride (ushorts)

typedef __attribute__((ext_vector_type(16))) __bf16 v16bf;
typedef __attribute__((ext_vector_type(8)))  float  v8f;

union Frag {
  v16bf bf;
  uint4 q[2];
  unsigned short u[16];
};

static __device__ inline unsigned short f2bf(float f) {
  unsigned int u = __float_as_uint(f);
  unsigned int r = u + 0x7FFFu + ((u >> 16) & 1u);   // round-to-nearest-even
  return (unsigned short)(r >> 16);
}
static __device__ inline float bf2f(unsigned short h) {
  return __uint_as_float(((unsigned int)h) << 16);
}
static __device__ inline v8f vzero8() {
  v8f z;
  #pragma unroll
  for (int i = 0; i < 8; ++i) z[i] = 0.0f;
  return z;
}

// A-matrix fragment (16x32 bf16, row-major source [M][ld]):
// lane<16: halves 0..7 -> K=kbase+0..7,  halves 8..15 -> K=kbase+16..23, row M=lane
// lane>=16: halves 0..7 -> K=kbase+8..15, halves 8..15 -> K=kbase+24..31, row M=lane-16
static __device__ inline Frag load_a_frag(const unsigned short* __restrict__ A,
                                          int row_base, int ld, int kbase, int lane) {
  Frag f;
  const int m  = row_base + (lane & 15);
  const int ko = kbase + ((lane & 16) ? 8 : 0);
  f.q[0] = *(const uint4*)(A + m * ld + ko);
  f.q[1] = *(const uint4*)(A + m * ld + ko + 16);
  return f;
}

// B-matrix fragment (32x16 bf16) from LDS row-major [spatial e][K contiguous]:
// lanes 0-15 hold K=kbase+0..7,16..23 (col N = lane), lanes 16-31 hold K=kbase+8..15,24..31
static __device__ inline Frag load_b_frag_lds(const unsigned short* S,
                                              int e, int ld, int kbase, int lane) {
  Frag f;
  const int ko = kbase + ((lane & 16) ? 8 : 0);
  const uint4* p = (const uint4*)(S + e * ld + ko);
  f.q[0] = p[0];
  f.q[1] = p[2];
  return f;
}

static __device__ inline v8f wmma_bf16(Frag a, Frag b, v8f c) {
  return __builtin_amdgcn_wmma_f32_16x16x32_bf16(false, a.bf, false, b.bf,
                                                 (short)0, c, false, false);
}

// CDNA5 async global->LDS copy (16B per lane), tracked by ASYNCcnt.
// dst must be a pointer into __shared__ (low 32 bits = wave-relative LDS offset).
static __device__ inline void async_copy_b128(void* lds_dst, const void* gsrc) {
  unsigned int ldsoff = (unsigned int)(uintptr_t)lds_dst;
  unsigned long long ga = (unsigned long long)(uintptr_t)gsrc;
  asm volatile("global_load_async_to_lds_b128 %0, %1, off"
               :: "v"(ldsoff), "v"(ga) : "memory");
}
static __device__ inline void wait_async0() {
  asm volatile("s_wait_asynccnt 0x0" ::: "memory");
}

// ---------------- weight prep: fp32 -> bf16, WMMA-friendly layouts ----------------
__global__ void k_prep(const float* __restrict__ w0, const float* __restrict__ w1,
                       const float* __restrict__ w2,
                       unsigned short* __restrict__ w0b,
                       unsigned short* __restrict__ w1b,
                       unsigned short* __restrict__ w2b) {
  const int i = blockIdx.x * 256 + threadIdx.x;
  if (i < 64 * 64) w0b[i] = f2bf(w0[i]);                       // [o][c], k=1 trailing
  if (i < 9 * OC1 * KCH) {                                     // w1b[t][o][c]
    const int t = i >> 15, rem = i & 32767, o = rem >> 6, c = rem & 63;
    w1b[i] = f2bf(w1[(o * 64 + c) * 9 + t]);
  }
  if (i < 3 * OC2 * IC2) {                                     // w2b[dh][co][i]
    const int dh = i >> 17, rem = i & 131071, co = rem >> 9, ii = rem & 511;
    w2b[i] = f2bf(w2[co * 1536 + dh * 512 + ii]);
  }
}

// ---------------- kernel 1: conv0(1x1) + conv1(k=9) + squash -> hsq bf16 ----------
__global__ __launch_bounds__(256)
void k_conv01(const float* __restrict__ x, const unsigned short* __restrict__ w0b,
              const float* __restrict__ b0, const unsigned short* __restrict__ w1b,
              const float* __restrict__ b1, unsigned short* __restrict__ hsq) {
  __shared__ unsigned short xs[EXT * XLD];  // x tile, transposed: [ext l][c]
  __shared__ unsigned short ys[EXT * XLD];  // conv0 out, transposed: [ext l][c]
  __shared__ float sqbuf[LT];

  const int tid   = threadIdx.x;
  const int lane  = tid & 31;
  const int wv    = tid >> 5;                  // 0..7
  const int b     = blockIdx.y;
  const int lbase = blockIdx.x * LT;
  const int hi16  = (lane & 16) ? 8 : 0;

  if (tid < LT) sqbuf[tid] = 0.0f;

  // stage x (coalesced in L), transpose into LDS, zero-pad halo
  for (int idx = tid; idx < KCH * EXT; idx += 256) {
    const int c = idx / EXT, e = idx - c * EXT;
    const int l = lbase - 8 + e;
    const float v = (l >= 0 && l < L_) ? x[(b * KCH + c) * L_ + l] : 0.0f;
    xs[e * XLD + c] = f2bf(v);
  }
  __syncthreads();

  // conv0: y0[64 x 80] = w0b[64x64] * xs ; 4 mtiles x 5 ntiles = 20 WMMA tiles
  for (int tix = wv; tix < 20; tix += 8) {
    const int mt = tix / 5, nt = tix - mt * 5;
    v8f acc = vzero8();
    #pragma unroll
    for (int kk = 0; kk < 64; kk += 32) {
      Frag a  = load_a_frag(w0b, mt * 16, 64, kk, lane);
      Frag bf = load_b_frag_lds(xs, nt * 16 + (lane & 15), XLD, kk, lane);
      acc = wmma_bf16(a, bf, acc);
    }
    const int col = nt * 16 + (lane & 15);
    const int mb  = mt * 16 + hi16;
    Frag o;
    #pragma unroll
    for (int r = 0; r < 8; ++r) o.u[r] = f2bf(acc[r] + b0[mb + r]);
    *(uint4*)&ys[col * XLD + mb] = o.q[0];
  }
  __syncthreads();

  // conv1: 512 outputs x 64 cols; wave -> 4 mtiles x 4 ntiles (16 accum tiles)
  v8f acc1[16];
  #pragma unroll
  for (int i = 0; i < 16; ++i) acc1[i] = vzero8();

  for (int t = 0; t < 9; ++t) {
    // pull next tap's A-panel toward L2 while this tap computes
    if (t < 8)
      __builtin_prefetch(w1b + (t + 1) * (OC1 * KCH) + (wv * 4) * 16 * KCH, 0, 0);
    #pragma unroll
    for (int kk2 = 0; kk2 < 2; ++kk2) {
      const int kk = kk2 * 32;
      Frag bfr[4];
      #pragma unroll
      for (int nt = 0; nt < 4; ++nt)
        bfr[nt] = load_b_frag_lds(ys, nt * 16 + (lane & 15) + t + 4, XLD, kk, lane);
      #pragma unroll
      for (int mi = 0; mi < 4; ++mi) {
        Frag a = load_a_frag(w1b + t * (OC1 * KCH), (wv * 4 + mi) * 16, KCH, kk, lane);
        #pragma unroll
        for (int nt = 0; nt < 4; ++nt)
          acc1[mi * 4 + nt] = wmma_bf16(a, bfr[nt], acc1[mi * 4 + nt]);
      }
    }
  }

  // bias + per-column sum of squares (cross-wave via LDS float atomics)
  #pragma unroll
  for (int mi = 0; mi < 4; ++mi) {
    const int mb = (wv * 4 + mi) * 16 + hi16;
    float bias[8];
    #pragma unroll
    for (int r = 0; r < 8; ++r) bias[r] = b1[mb + r];
    #pragma unroll
    for (int nt = 0; nt < 4; ++nt) {
      float p = 0.0f;
      #pragma unroll
      for (int r = 0; r < 8; ++r) {
        const float v = acc1[mi * 4 + nt][r] + bias[r];
        acc1[mi * 4 + nt][r] = v;
        p += v * v;
      }
      atomicAdd(&sqbuf[nt * 16 + (lane & 15)], p);
    }
  }
  __syncthreads();

  // squash columns, store bf16 to global (16B per lane per tile)
  #pragma unroll
  for (int nt = 0; nt < 4; ++nt) {
    const int c0 = nt * 16 + (lane & 15);
    const float sq = sqbuf[c0];
    const float scale = (sq / (1.0f + sq)) * rsqrtf(sq + 1e-8f);
    const size_t rowoff = ((size_t)b * L_ + (lbase + c0)) * OC1;
    #pragma unroll
    for (int mi = 0; mi < 4; ++mi) {
      const int mb = (wv * 4 + mi) * 16 + hi16;
      Frag o;
      #pragma unroll
      for (int r = 0; r < 8; ++r) o.u[r] = f2bf(acc1[mi * 4 + nt][r] * scale);
      *(uint4*)&hsq[rowoff + mb] = o.q[0];
    }
  }
}

// ---------------- kernel 2: conv2 (dilated-8 GEMM) + dynamic routing -------------
__global__ __launch_bounds__(256)
void k_conv2_route(const unsigned short* __restrict__ hsq,
                   const unsigned short* __restrict__ w2b,
                   const float* __restrict__ b2, float* __restrict__ out) {
  __shared__ unsigned short hsb[EXT * XLD];   // staged hsq chunk [ext l][64 i]
  __shared__ unsigned short vt[LT * VLD];     // V tile bf16 [col][co]
  __shared__ float denom[8];

  const int tid   = threadIdx.x;
  const int lane  = tid & 31;
  const int wv    = tid >> 5;
  const int b     = blockIdx.y;
  const int lbase = blockIdx.x * LT;
  const int hi16  = (lane & 16) ? 8 : 0;

  v8f acc[8];                                  // 2 mtiles x 4 ntiles per wave
  #pragma unroll
  for (int i = 0; i < 8; ++i) acc[i] = vzero8();

  for (int ic = 0; ic < 8; ++ic) {             // i-chunks of 64
    __syncthreads();
    // async global->LDS staging (16B per lane), zero-fill halo rows
    for (int idx = tid; idx < EXT * 8; idx += 256) {   // 8 x uint4 per row
      const int e = idx >> 3, qi = idx & 7;
      const int l = lbase - 8 + e;
      unsigned short* dst = &hsb[e * XLD + qi * 8];
      if (l >= 0 && l < L_) {
        async_copy_b128(dst, &hsq[((size_t)b * L_ + l) * IC2 + ic * 64 + qi * 8]);
      } else {
        *(uint4*)dst = make_uint4(0u, 0u, 0u, 0u);
      }
    }
    wait_async0();
    __syncthreads();

    #pragma unroll
    for (int dh = 0; dh < 3; ++dh) {
      #pragma unroll
      for (int kk2 = 0; kk2 < 2; ++kk2) {
        const int kk = kk2 * 32;
        Frag bfr[4];
        #pragma unroll
        for (int nt = 0; nt < 4; ++nt)
          bfr[nt] = load_b_frag_lds(hsb, nt * 16 + (lane & 15) + dh * 8, XLD, kk, lane);
        #pragma unroll
        for (int mi = 0; mi < 2; ++mi) {
          Frag a = load_a_frag(w2b + dh * (OC2 * IC2), (wv * 2 + mi) * 16,
                               IC2, ic * 64 + kk, lane);
          #pragma unroll
          for (int nt = 0; nt < 4; ++nt)
            acc[mi * 4 + nt] = wmma_bf16(a, bfr[nt], acc[mi * 4 + nt]);
        }
      }
    }
  }
  __syncthreads();

  // bias, stash V (bf16) to LDS: vt[col][co]
  #pragma unroll
  for (int mi = 0; mi < 2; ++mi) {
    const int mb = (wv * 2 + mi) * 16 + hi16;
    #pragma unroll
    for (int nt = 0; nt < 4; ++nt) {
      const int col = nt * 16 + (lane & 15);
      Frag o;
      #pragma unroll
      for (int r = 0; r < 8; ++r) o.u[r] = f2bf(acc[mi * 4 + nt][r] + b2[mb + r]);
      *(uint4*)&vt[col * VLD + mb] = o.q[0];
    }
  }
  __syncthreads();

  // dynamic routing: 8 capsule groups (ln) per tile; thread = (csb, asb)
  const int csb = (tid >> 4) & 15;
  const int asb = tid & 15;
  for (int g = 0; g < 8; ++g) {
    float V[8];
    #pragma unroll
    for (int n = 0; n < 8; ++n) V[n] = bf2f(vt[(g * 8 + n) * VLD + csb * 16 + asb]);
    float blog[8];
    #pragma unroll
    for (int n = 0; n < 8; ++n) blog[n] = 0.0f;
    float vout = 0.0f;

    for (int it = 0; it < 3; ++it) {
      __syncthreads();
      if (tid < 8) denom[tid] = 0.0f;
      __syncthreads();
      float e[8];
      #pragma unroll
      for (int n = 0; n < 8; ++n) e[n] = __expf(blog[n]);
      #pragma unroll
      for (int n = 0; n < 8; ++n)                      // sum over csb, per n
        if (asb == n) atomicAdd(&denom[n], e[n]);
      __syncthreads();
      float s = 0.0f;
      #pragma unroll
      for (int n = 0; n < 8; ++n) s += (e[n] / denom[n]) * V[n];
      float sq = s * s;                                // reduce over asb (lane&15)
      sq += __shfl_xor(sq, 1); sq += __shfl_xor(sq, 2);
      sq += __shfl_xor(sq, 4); sq += __shfl_xor(sq, 8);
      const float v = s * (sq / (1.0f + sq)) * rsqrtf(sq + 1e-8f);
      vout = v;
      if (it < 2) {
        #pragma unroll
        for (int n = 0; n < 8; ++n) {
          float a = V[n] * v;
          a += __shfl_xor(a, 1); a += __shfl_xor(a, 2);
          a += __shfl_xor(a, 4); a += __shfl_xor(a, 8);
          blog[n] += a;
        }
      }
    }
    const int ln = blockIdx.x * 8 + g;
    out[(((size_t)b * 256 + ln) * 16 + csb) * 16 + asb] = vout;
  }
}

// ---------------- launch ----------------
// ws layout: w0b 8KB | w1b 576KB | w2b 768KB | hsq 64MB  (total ~65.4 MB)
extern "C" void kernel_launch(void* const* d_in, const int* in_sizes, int n_in,
                              void* d_out, int out_size, void* d_ws, size_t ws_size,
                              hipStream_t stream) {
  const float* x  = (const float*)d_in[0];
  const float* w0 = (const float*)d_in[1];
  const float* b0 = (const float*)d_in[2];
  const float* w1 = (const float*)d_in[3];
  const float* b1 = (const float*)d_in[4];
  const float* w2 = (const float*)d_in[5];
  const float* b2 = (const float*)d_in[6];
  float* out = (float*)d_out;

  char* ws = (char*)d_ws;
  unsigned short* w0b = (unsigned short*)(ws);
  unsigned short* w1b = (unsigned short*)(ws + 8192);
  unsigned short* w2b = (unsigned short*)(ws + 8192 + 589824);
  unsigned short* hsq = (unsigned short*)(ws + 1384448);

  k_prep<<<1536, 256, 0, stream>>>(w0, w1, w2, w0b, w1b, w2b);
  dim3 grid(32, 32);
  k_conv01<<<grid, 256, 0, stream>>>(x, w0b, b0, w1b, b1, hsq);
  k_conv2_route<<<grid, 256, 0, stream>>>(hsq, w2b, b2, out);
}
